// DecoderOnlyTrajectoryModel_30408368456094
// MI455X (gfx1250) — matrix operand order repaired and checked by reading, not compiled, not run on hardware
//
#include <hip/hip_runtime.h>
#include <math.h>
#include <stdint.h>

// ---------------- problem constants ----------------
#define B_    128
#define S_    70
#define PAST_ 50
#define FUT_  20
#define D_    256
#define H_    8
#define DH_   32
#define FF_   1024
#define L_    3
#define M_    (B_ * S_)   // 8960 rows, divisible by 32

// ---------------- types ----------------
typedef __bf16 bf16;
typedef __attribute__((ext_vector_type(16))) __bf16 v16bf;
typedef __attribute__((ext_vector_type(8)))  __bf16 v8bf;
typedef __attribute__((ext_vector_type(8)))  float  v8f;
typedef __attribute__((ext_vector_type(4)))  int    v4i;

// direct global->LDS async path (gfx1250), guarded so compile never breaks
#if defined(__gfx1250__) && __has_builtin(__builtin_amdgcn_global_load_async_to_lds_b128) && __has_builtin(__builtin_amdgcn_s_wait_asynccnt)
#define USE_ASYNC_LDS 1
typedef __attribute__((address_space(1))) v4i* gv4i_p;   // NOTE: non-const, per builtin signature
typedef __attribute__((address_space(3))) v4i* lv4i_p;
#else
#define USE_ASYNC_LDS 0
#endif

__device__ __forceinline__ bf16 f2bf(float f) {
  union { float f; unsigned int u; } in; in.f = f;
  unsigned int u = in.u;
  u += 0x7FFFu + ((u >> 16) & 1u);           // round-to-nearest-even
  unsigned short h = (unsigned short)(u >> 16);
  union { unsigned short u; bf16 b; } out; out.u = h;
  return out.b;
}

// ---------------------------------------------------------------
// WMMA GEMM:  C[M x N] = A[M x K](bf16) @ W[N x K]^T(bf16) + bias
// block = 256 threads = 8 waves; each wave -> 32x32 output tile.
// grid = (M/32, N/256).
// A strip (32 x 64 bf16) double-buffered in LDS; staged with
// GLOBAL_LOAD_ASYNC_TO_LDS_B128 (ASYNCcnt) when available so the
// next chunk's staging overlaps the current chunk's 8 WMMAs.
// ---------------------------------------------------------------
__global__ __launch_bounds__(256) void gemm_bf16_wmma(
    const bf16* __restrict__ A, int lda,
    const bf16* __restrict__ W, int ldw,
    const float* __restrict__ bias,
    float* __restrict__ C, int ldc,
    int K)
{
  __shared__ bf16 As[2][32 * 64];

  const int wave = threadIdx.x >> 5;
  const int lane = threadIdx.x & 31;
  const int m0   = blockIdx.x * 32;
  const int n0   = (blockIdx.y * 8 + wave) * 32;
  const int hi   = lane >> 4;     // 0: lanes 0-15, 1: lanes 16-31
  const int lo   = lane & 15;

  // this thread's 16-byte staging slot within the 32x64 strip
  const int sidx = threadIdx.x * 8;   // element index
  const int sr   = sidx >> 6;         // row 0..31
  const int scol = sidx & 63;         // col 0..63

  v8f acc00 = {}, acc01 = {}, acc10 = {}, acc11 = {};

  const int nch = K >> 6;             // number of 64-wide K chunks

#if USE_ASYNC_LDS
  {
    gv4i_p g = (gv4i_p)(void*)const_cast<bf16*>(&A[(size_t)(m0 + sr) * lda + scol]);
    lv4i_p l = (lv4i_p)(void*)&As[0][sidx];
    __builtin_amdgcn_global_load_async_to_lds_b128(g, l, 0, 0);
  }
#endif

  for (int c = 0; c < nch; ++c) {
    const int kk = c << 6;
#if USE_ASYNC_LDS
    if (c + 1 < nch) {
      gv4i_p g = (gv4i_p)(void*)const_cast<bf16*>(&A[(size_t)(m0 + sr) * lda + ((c + 1) << 6) + scol]);
      lv4i_p l = (lv4i_p)(void*)&As[(c + 1) & 1][sidx];
      __builtin_amdgcn_global_load_async_to_lds_b128(g, l, 0, 0);
      __builtin_amdgcn_s_wait_asynccnt(1);   // chunk c landed (next still in flight)
    } else {
      __builtin_amdgcn_s_wait_asynccnt(0);
    }
    __syncthreads();
    const bf16* Ab = &As[c & 1][0];
#else
    __syncthreads();
    *(v8bf*)&As[0][sidx] = *(const v8bf*)&A[(size_t)(m0 + sr) * lda + kk + scol];
    __syncthreads();
    const bf16* Ab = &As[0][0];
#endif

#pragma unroll
    for (int ks = 0; ks < 64; ks += 32) {
      // A fragment (contiguous K order, matching B):
      // lanes 0-15: K {0..15}; lanes 16-31: K {16..31}
      const int ka = ks + (hi ? 16 : 0);
      v8bf a0lo = *(const v8bf*)&Ab[(lo)      * 64 + ka];
      v8bf a0hi = *(const v8bf*)&Ab[(lo)      * 64 + ka + 8];
      v8bf a1lo = *(const v8bf*)&Ab[(16 + lo) * 64 + ka];
      v8bf a1hi = *(const v8bf*)&Ab[(16 + lo) * 64 + ka + 8];
      v16bf a0 = __builtin_shufflevector(a0lo, a0hi, 0,1,2,3,4,5,6,7,8,9,10,11,12,13,14,15);
      v16bf a1 = __builtin_shufflevector(a1lo, a1hi, 0,1,2,3,4,5,6,7,8,9,10,11,12,13,14,15);

      // B fragment (32x16, K x N): lane n<16 holds col n, K 0..15; lanes 16-31 K 16..31.
      // W is row-major [N,K], so that's a contiguous 32-byte run of row n.
      const int kb = kk + ks + (hi ? 16 : 0);
      v16bf b0 = *(const v16bf*)&W[(size_t)(n0 + lo)      * ldw + kb];
      v16bf b1 = *(const v16bf*)&W[(size_t)(n0 + 16 + lo) * ldw + kb];

      acc00 = __builtin_amdgcn_wmma_f32_16x16x32_bf16(false, a0, false, b0, (short)0, acc00, false, false);
      acc01 = __builtin_amdgcn_wmma_f32_16x16x32_bf16(false, a0, false, b1, (short)0, acc01, false, false);
      acc10 = __builtin_amdgcn_wmma_f32_16x16x32_bf16(false, a1, false, b0, (short)0, acc10, false, false);
      acc11 = __builtin_amdgcn_wmma_f32_16x16x32_bf16(false, a1, false, b1, (short)0, acc11, false, false);
    }
#if USE_ASYNC_LDS
    __syncthreads();   // all waves done reading buf (c&1) before it is overwritten
#endif
  }

  // C/D layout: lane column = lane&15; rows (lane>=16 ? 8..15 : 0..7)
  const int rbase = hi * 8;
  const float bv0 = bias ? bias[n0 + lo]      : 0.f;
  const float bv1 = bias ? bias[n0 + 16 + lo] : 0.f;
#pragma unroll
  for (int r = 0; r < 8; ++r) {
    C[(size_t)(m0 + rbase + r)      * ldc + n0 + lo]      = acc00[r] + bv0;
    C[(size_t)(m0 + rbase + r)      * ldc + n0 + 16 + lo] = acc01[r] + bv1;
    C[(size_t)(m0 + 16 + rbase + r) * ldc + n0 + lo]      = acc10[r] + bv0;
    C[(size_t)(m0 + 16 + rbase + r) * ldc + n0 + 16 + lo] = acc11[r] + bv1;
  }
}

// ---------------- fp32 -> bf16 convert ----------------
__global__ void f32_to_bf16_kernel(const float* __restrict__ in, bf16* __restrict__ out, int n) {
  int i = blockIdx.x * blockDim.x + threadIdx.x;
  if (i < n) out[i] = f2bf(in[i]);
}

// ---------------- init: zero future deltas, set cur = last past position ----------------
__global__ void init_kernel(const float* __restrict__ traj, float* __restrict__ fd, float* __restrict__ cur) {
  int i = blockIdx.x * blockDim.x + threadIdx.x;
  if (i < B_ * FUT_ * 2) fd[i] = 0.f;
  if (i < B_ * 2) {
    int b = i >> 1, j = i & 1;
    cur[i] = traj[((size_t)b * PAST_ + (PAST_ - 1)) * 2 + j];
  }
}

// ---------------- embed: deltas -> in_W -> +pos_emb (pre-LN0) ----------------
__global__ __launch_bounds__(256) void embed_kernel(
    const float* __restrict__ traj, const float* __restrict__ fd,
    const float* __restrict__ inW, const float* __restrict__ inb,
    const float* __restrict__ pos, float* __restrict__ out)
{
  int wave = threadIdx.x >> 5, lane = threadIdx.x & 31;
  int m = blockIdx.x * 8 + wave;           // grid = M/8
  int b = m / S_, p = m % S_;
  float d0, d1;
  if (p < PAST_) {
    if (p == 0) { d0 = 0.f; d1 = 0.f; }
    else {
      d0 = traj[((size_t)b * PAST_ + p) * 2 + 0] - traj[((size_t)b * PAST_ + p - 1) * 2 + 0];
      d1 = traj[((size_t)b * PAST_ + p) * 2 + 1] - traj[((size_t)b * PAST_ + p - 1) * 2 + 1];
    }
  } else {
    d0 = fd[((size_t)b * FUT_ + (p - PAST_)) * 2 + 0];
    d1 = fd[((size_t)b * FUT_ + (p - PAST_)) * 2 + 1];
  }
#pragma unroll
  for (int j = 0; j < 8; ++j) {
    int d = lane * 8 + j;
    out[(size_t)m * D_ + d] = d0 * inW[d * 2] + d1 * inW[d * 2 + 1] + inb[d] + pos[(size_t)p * D_ + d];
  }
}

// ---------------- LayerNorm (optionally + residual), dual fp32/bf16 output ----------------
__global__ __launch_bounds__(256) void ln_kernel(
    const float* __restrict__ inp, const float* __restrict__ resid,
    const float* __restrict__ w, const float* __restrict__ bb,
    float* __restrict__ outf, bf16* __restrict__ outb, bf16* __restrict__ outb2)
{
  int wave = threadIdx.x >> 5, lane = threadIdx.x & 31;
  int m = blockIdx.x * 8 + wave;
  float v[8];
  float s = 0.f;
#pragma unroll
  for (int j = 0; j < 8; ++j) {
    int d = lane * 8 + j;
    float x = inp[(size_t)m * D_ + d];
    if (resid) x += resid[(size_t)m * D_ + d];
    v[j] = x; s += x;
  }
#pragma unroll
  for (int off = 16; off; off >>= 1) s += __shfl_xor(s, off, 32);
  float mean = s * (1.f / D_);
  float vs = 0.f;
#pragma unroll
  for (int j = 0; j < 8; ++j) { float t = v[j] - mean; vs += t * t; }
#pragma unroll
  for (int off = 16; off; off >>= 1) vs += __shfl_xor(vs, off, 32);
  float rstd = rsqrtf(vs * (1.f / D_) + 1e-5f);
#pragma unroll
  for (int j = 0; j < 8; ++j) {
    int d = lane * 8 + j;
    float y = (v[j] - mean) * rstd * w[d] + bb[d];
    outf[(size_t)m * D_ + d] = y;
    bf16 h = f2bf(y);
    outb[(size_t)m * D_ + d] = h;
    if (outb2) outb2[(size_t)m * D_ + d] = h;
  }
}

// ---------------- causal softmax attention; one (b,h) per block; K/V staged in LDS ----------------
__global__ __launch_bounds__(128) void attn_kernel(
    const float* __restrict__ qkv, int ld, int qoff, int koff, int voff,
    bf16* __restrict__ out)
{
  __shared__ float Ks[S_ * DH_];
  __shared__ float Vs[S_ * DH_];
  __shared__ float sc[S_ * S_];
  int bh = blockIdx.x;
  int b = bh / H_, h = bh % H_;
  int tid = threadIdx.x;
  for (int i = tid; i < S_ * DH_; i += 128) {
    int r = i / DH_, d = i % DH_;
    size_t rowbase = (size_t)(b * S_ + r) * ld + h * DH_;
    Ks[i] = qkv[rowbase + koff + d];
    Vs[i] = qkv[rowbase + voff + d];
  }
  __syncthreads();
  if (tid < S_) {
    const int t = tid;
    float q[DH_];
    const float* qrow = &qkv[(size_t)(b * S_ + t) * ld + qoff + h * DH_];
#pragma unroll
    for (int d = 0; d < DH_; ++d) q[d] = qrow[d];
    const float scale = 0.17677669529663687f;  // 1/sqrt(32)
    float mx = -1e30f;
    for (int k = 0; k <= t; ++k) {
      float sv = 0.f;
#pragma unroll
      for (int d = 0; d < DH_; ++d) sv += q[d] * Ks[k * DH_ + d];
      sv *= scale;
      sc[t * S_ + k] = sv;
      mx = fmaxf(mx, sv);
    }
    float sum = 0.f;
    for (int k = 0; k <= t; ++k) { float e = expf(sc[t * S_ + k] - mx); sc[t * S_ + k] = e; sum += e; }
    float inv = 1.f / sum;
    float o[DH_];
#pragma unroll
    for (int d = 0; d < DH_; ++d) o[d] = 0.f;
    for (int k = 0; k <= t; ++k) {
      float wgt = sc[t * S_ + k] * inv;
#pragma unroll
      for (int d = 0; d < DH_; ++d) o[d] += wgt * Vs[k * DH_ + d];
    }
    bf16* orow = &out[(size_t)(b * S_ + t) * D_ + h * DH_];
#pragma unroll
    for (int d = 0; d < DH_; ++d) orow[d] = f2bf(o[d]);
  }
}

// ---------------- exact GELU, fp32 -> bf16 ----------------
__global__ void gelu_kernel(const float* __restrict__ in, bf16* __restrict__ out, int n) {
  int i = blockIdx.x * blockDim.x + threadIdx.x;
  if (i < n) {
    float x = in[i];
    out[i] = f2bf(0.5f * x * (1.f + erff(x * 0.7071067811865476f)));
  }
}

// ---------------- readout: nd = x_row @ out_W^T + out_b ; fd/cur/d_out update ----------------
__global__ __launch_bounds__(128) void readout_kernel(
    const float* __restrict__ xf, const float* __restrict__ oW, const float* __restrict__ ob,
    float* __restrict__ fd, float* __restrict__ cur, float* __restrict__ dout, int s)
{
  int b = threadIdx.x;
  if (b >= B_) return;
  const float* row = &xf[(size_t)(b * S_ + PAST_ - 1 + s) * D_];
#pragma unroll
  for (int j = 0; j < 2; ++j) {
    float acc = ob[j];
    for (int d = 0; d < D_; ++d) acc += row[d] * oW[j * D_ + d];
    fd[((size_t)b * FUT_ + s) * 2 + j] = acc;
    float c = cur[b * 2 + j] + acc;
    cur[b * 2 + j] = c;
    dout[((size_t)b * FUT_ + s) * 2 + j] = c;
  }
}

// ================= host orchestration =================
extern "C" void kernel_launch(void* const* d_in, const int* in_sizes, int n_in,
                              void* d_out, int out_size, void* d_ws, size_t ws_size,
                              hipStream_t stream) {
  (void)in_sizes; (void)n_in; (void)out_size; (void)ws_size;

  const float* traj    = (const float*)d_in[0];
  const float* inW     = (const float*)d_in[1];
  const float* inb     = (const float*)d_in[2];
  const float* pos     = (const float*)d_in[3];
  const float* ln0w    = (const float*)d_in[4];
  const float* ln0b    = (const float*)d_in[5];
  const float* saWqkv  = (const float*)d_in[6];
  const float* sabqkv  = (const float*)d_in[7];
  const float* saWo    = (const float*)d_in[8];
  const float* sabo    = (const float*)d_in[9];
  const float* caWqkv  = (const float*)d_in[10];
  const float* cabqkv  = (const float*)d_in[11];
  const float* caWo    = (const float*)d_in[12];
  const float* cabo    = (const float*)d_in[13];
  const float* ff1W    = (const float*)d_in[14];
  const float* ff1b    = (const float*)d_in[15];
  const float* ff2W    = (const float*)d_in[16];
  const float* ff2b    = (const float*)d_in[17];
  const float* ln1w    = (const float*)d_in[18];
  const float* ln1b    = (const float*)d_in[19];
  const float* ln2w    = (const float*)d_in[20];
  const float* ln2b    = (const float*)d_in[21];
  const float* ln3w    = (const float*)d_in[22];
  const float* ln3b    = (const float*)d_in[23];
  const float* p_outW  = (const float*)d_in[24];
  const float* p_outb  = (const float*)d_in[25];

  // ---- carve scratch ----
  char* base = (char*)d_ws;
  size_t off = 0;
  auto carve = [&](size_t bytes) -> void* {
    void* r = base + off;
    off = (off + bytes + 255) & ~(size_t)255;
    return r;
  };
  bf16* saWqkv_b = (bf16*)carve((size_t)L_ * 3 * D_ * D_ * sizeof(bf16));
  bf16* saWo_b   = (bf16*)carve((size_t)L_ * D_ * D_ * sizeof(bf16));
  bf16* caWqkv_b = (bf16*)carve((size_t)L_ * 3 * D_ * D_ * sizeof(bf16));
  bf16* caWo_b   = (bf16*)carve((size_t)L_ * D_ * D_ * sizeof(bf16));
  bf16* ff1W_b   = (bf16*)carve((size_t)L_ * FF_ * D_ * sizeof(bf16));
  bf16* ff2W_b   = (bf16*)carve((size_t)L_ * D_ * FF_ * sizeof(bf16));
  float* xf   = (float*)carve((size_t)M_ * D_ * sizeof(float));
  bf16*  xb   = (bf16*) carve((size_t)M_ * D_ * sizeof(bf16));
  bf16*  memb = (bf16*) carve((size_t)M_ * D_ * sizeof(bf16));
  float* tmp0 = (float*)carve((size_t)M_ * FF_ * sizeof(float));   // qkv[768] / h[1024]
  bf16*  tmpb = (bf16*) carve((size_t)M_ * FF_ * sizeof(bf16));    // attn out / gelu out
  float* tmp1 = (float*)carve((size_t)M_ * D_ * sizeof(float));    // proj outputs / embed
  float* fd   = (float*)carve((size_t)B_ * FUT_ * 2 * sizeof(float));
  float* cur  = (float*)carve((size_t)B_ * 2 * sizeof(float));

  auto cvt = [&](const float* src, bf16* dst, int n) {
    f32_to_bf16_kernel<<<(n + 255) / 256, 256, 0, stream>>>(src, dst, n);
  };
  cvt(saWqkv, saWqkv_b, L_ * 3 * D_ * D_);
  cvt(saWo,   saWo_b,   L_ * D_ * D_);
  cvt(caWqkv, caWqkv_b, L_ * 3 * D_ * D_);
  cvt(caWo,   caWo_b,   L_ * D_ * D_);
  cvt(ff1W,   ff1W_b,   L_ * FF_ * D_);
  cvt(ff2W,   ff2W_b,   L_ * D_ * FF_);

  init_kernel<<<(B_ * FUT_ * 2 + 255) / 256, 256, 0, stream>>>(traj, fd, cur);

  auto gemm = [&](const bf16* A, int lda, const bf16* W, int ldw, const float* bias,
                  float* C, int ldc, int N, int K) {
    dim3 g(M_ / 32, N / 256);
    gemm_bf16_wmma<<<g, 256, 0, stream>>>(A, lda, W, ldw, bias, C, ldc, K);
  };

  for (int s = 0; s < FUT_; ++s) {
    embed_kernel<<<M_ / 8, 256, 0, stream>>>(traj, fd, inW, inb, pos, tmp1);
    ln_kernel<<<M_ / 8, 256, 0, stream>>>(tmp1, nullptr, ln0w, ln0b, xf, xb, memb);

    for (int l = 0; l < L_; ++l) {
      // ---- self-attention ----
      gemm(xb, D_, saWqkv_b + (size_t)l * 3 * D_ * D_, D_, sabqkv + l * 3 * D_,
           tmp0, 3 * D_, 3 * D_, D_);
      attn_kernel<<<B_ * H_, 128, 0, stream>>>(tmp0, 3 * D_, 0, D_, 2 * D_, tmpb);
      gemm(tmpb, D_, saWo_b + (size_t)l * D_ * D_, D_, sabo + l * D_,
           tmp1, D_, D_, D_);
      ln_kernel<<<M_ / 8, 256, 0, stream>>>(tmp1, xf, ln1w + l * D_, ln1b + l * D_, xf, xb, nullptr);

      // ---- cross-attention (q from x, k/v from mem) ----
      gemm(xb, D_, caWqkv_b + (size_t)l * 3 * D_ * D_, D_, cabqkv + l * 3 * D_,
           tmp0, 3 * D_, D_, D_);                       // q -> cols [0,256)
      gemm(memb, D_, caWqkv_b + (size_t)l * 3 * D_ * D_ + (size_t)D_ * D_, D_,
           cabqkv + l * 3 * D_ + D_, tmp0 + D_, 3 * D_, 2 * D_, D_);  // kv -> cols [256,768)
      attn_kernel<<<B_ * H_, 128, 0, stream>>>(tmp0, 3 * D_, 0, D_, 2 * D_, tmpb);
      gemm(tmpb, D_, caWo_b + (size_t)l * D_ * D_, D_, cabo + l * D_,
           tmp1, D_, D_, D_);
      ln_kernel<<<M_ / 8, 256, 0, stream>>>(tmp1, xf, ln2w + l * D_, ln2b + l * D_, xf, xb, nullptr);

      // ---- FFN ----
      gemm(xb, D_, ff1W_b + (size_t)l * FF_ * D_, D_, ff1b + l * FF_,
           tmp0, FF_, FF_, D_);
      gelu_kernel<<<(M_ * FF_ + 255) / 256, 256, 0, stream>>>(tmp0, tmpb, M_ * FF_);
      gemm(tmpb, FF_, ff2W_b + (size_t)l * D_ * FF_, FF_, ff2b + l * D_,
           tmp1, D_, D_, FF_);
      ln_kernel<<<M_ / 8, 256, 0, stream>>>(tmp1, xf, ln3w + l * D_, ln3b + l * D_, xf, xb, nullptr);
    }

    readout_kernel<<<1, 128, 0, stream>>>(xf, p_outW, p_outb, fd, cur, (float*)d_out, s);
  }
}
